// MambaLayer_14431090115013
// MI455X (gfx1250) — matrix operand order repaired and checked by reading, not compiled, not run on hardware
//
#include <hip/hip_runtime.h>

// ---------------------------------------------------------------------------
// Mamba layer for MI455X (gfx1250, wave32, WMMA).
// B=8, L=4096, D_MODEL=512, D_INNER=1024, D_STATE=16, D_CONV=4, DT_RANK=32
// GEMMs: bf16 x bf16 -> f32 via v_wmma_f32_16x16x32_bf16, LDS tiles staged
// with global_load_async_to_lds_b128 (ASYNCcnt path). Scan keeps fp32 state
// in registers with fused D-skip + SiLU gate epilogue.
// ---------------------------------------------------------------------------

typedef __attribute__((ext_vector_type(16))) __bf16 v16bf;
typedef __attribute__((ext_vector_type(8)))  float  v8f;

union Frag { v16bf v; unsigned int u[8]; };

#define NTOK   32768        // B*L
#define DMODEL 512
#define DINNER 1024
#define DSTATE 16

// ---- workspace layout (bytes) ----
#define OFF_XN     ((size_t)0)            //  xn bf16   [32768,512]   32 MB
#define OFF_XZ     ((size_t)33554432)     //  xz bf16   [32768,2048] 128 MB
#define OFF_UC     ((size_t)167772160)    //  uc bf16   [32768,1024]  64 MB
#define OFF_XDBLF  ((size_t)234881024)    //  x_dbl f32 [32768,64]     8 MB
#define OFF_XDBLB  ((size_t)243269632)    //  x_dbl bf16[32768,64]     4 MB
#define OFF_DT     ((size_t)247463936)    //  dt f32    [32768,1024] 128 MB
#define OFF_YG     ((size_t)381681664)    //  y*gate bf16 [32768,1024] 64 MB
#define OFF_WIN    ((size_t)448790528)    //  W_in bf16
#define OFF_WXP    ((size_t)450887680)    //  W_xp bf16
#define OFF_WDT    ((size_t)451018752)    //  W_dt bf16
#define OFF_WOUT   ((size_t)451084288)    //  W_out bf16
#define OFF_ANEG   ((size_t)452132864)    //  -exp(A_log) f32

// ---------------------------------------------------------------------------
// CDNA5 async global->LDS copy (16B per lane), tracked by ASYNCcnt.
// ---------------------------------------------------------------------------
__device__ __forceinline__ void async_ld16(unsigned lds_off, const void* gp)
{
    asm volatile("global_load_async_to_lds_b128 %0, %1, off"
                 :: "v"(lds_off), "v"(gp) : "memory");
}
__device__ __forceinline__ void wait_async0()
{
    asm volatile("s_wait_asynccnt 0" ::: "memory");
}

// ---------------------------------------------------------------------------
// Prep: convert weights to bf16, compute A = -exp(A_log)
// ---------------------------------------------------------------------------
__global__ __launch_bounds__(256)
void prep_kernel(const float* __restrict__ Win, const float* __restrict__ Wxp,
                 const float* __restrict__ Wdt, const float* __restrict__ Wout,
                 const float* __restrict__ Alog,
                 __bf16* __restrict__ bWin, __bf16* __restrict__ bWxp,
                 __bf16* __restrict__ bWdt, __bf16* __restrict__ bWout,
                 float* __restrict__ Aneg)
{
    int i = blockIdx.x * 256 + threadIdx.x;
    if (i < 1048576)        bWin[i]             = (__bf16)Win[i];
    else if (i < 1114112)   bWxp[i - 1048576]   = (__bf16)Wxp[i - 1048576];
    else if (i < 1146880)   bWdt[i - 1114112]   = (__bf16)Wdt[i - 1114112];
    else if (i < 1671168)   bWout[i - 1146880]  = (__bf16)Wout[i - 1146880];
    else if (i < 1687552) { int j = i - 1671168; Aneg[j] = -__expf(Alog[j]); }
}

// ---------------------------------------------------------------------------
// LayerNorm over D_MODEL=512, one token per block, output bf16
// ---------------------------------------------------------------------------
__global__ __launch_bounds__(256)
void layernorm_kernel(const float* __restrict__ x, const float* __restrict__ g,
                      const float* __restrict__ bta, __bf16* __restrict__ xn)
{
    const int t = threadIdx.x;
    const size_t base = (size_t)blockIdx.x * DMODEL;
    float v0 = x[base + t], v1 = x[base + t + 256];
    __shared__ float rs[256], rq[256];
    rs[t] = v0 + v1;
    rq[t] = v0 * v0 + v1 * v1;
    __syncthreads();
    #pragma unroll
    for (int o = 128; o > 0; o >>= 1) {
        if (t < o) { rs[t] += rs[t + o]; rq[t] += rq[t + o]; }
        __syncthreads();
    }
    float mu  = rs[0] * (1.f / 512.f);
    float var = rq[0] * (1.f / 512.f) - mu * mu;
    float ri  = rsqrtf(var + 1e-5f);
    xn[base + t]       = (__bf16)((v0 - mu) * ri * g[t]       + bta[t]);
    xn[base + t + 256] = (__bf16)((v1 - mu) * ri * g[t + 256] + bta[t + 256]);
}

// ---------------------------------------------------------------------------
// WMMA GEMM: C[M,N] = A[M,K] * W[N,K]^T, bf16 inputs, f32 accumulate.
// Block tile 128x64, 256 threads = 8 waves (wave32), wave tile 32x32
// (2 A frags x 2 B frags -> 4 WMMAs per K-step, fragments reused).
// LDS tiles staged via global_load_async_to_lds_b128 (no VGPR round-trip).
// Epilogue MODE: 0 = store bf16
//               1 = +bias, softplus, store f32           (dt projection)
//               2 = store f32 AND bf16                   (x_dbl)
//               3 = +residual, store f32                 (output proj)
// ---------------------------------------------------------------------------
template<int MODE>
__global__ __launch_bounds__(256)
void wmma_gemm_kernel(const __bf16* __restrict__ A, const __bf16* __restrict__ W,
                      int lda, int ldw, int K, int ldo,
                      float* __restrict__ outF, __bf16* __restrict__ outB,
                      const float* __restrict__ bias, const float* __restrict__ resid)
{
    __shared__ __align__(16) __bf16 sA[128 * 32];   // 8 KB
    __shared__ __align__(16) __bf16 sW[64 * 32];    // 4 KB

    const int t     = threadIdx.x;
    const int lane  = t & 31;
    const int wave  = t >> 5;
    const int waveM = wave >> 1;   // 0..3
    const int waveN = wave & 1;    // 0..1
    const int hi    = lane >> 4;   // lane half
    const int lr    = lane & 15;

    const size_t blockM = (size_t)blockIdx.y * 128;
    const int    blockN = blockIdx.x * 64;

    // Cooperative staging: 3 x 16B per thread (A rows 0..63, A rows 64..127, W)
    const int arow0 = t >> 2;              // 0..63
    const int arow1 = arow0 + 64;          // 64..127
    const int acol  = (t & 3) * 8;         // 0,8,16,24
    const unsigned sA0 = (unsigned)(uintptr_t)&sA[arow0 * 32 + acol];
    const unsigned sA1 = (unsigned)(uintptr_t)&sA[arow1 * 32 + acol];
    const unsigned sW0 = (unsigned)(uintptr_t)&sW[arow0 * 32 + acol];

    v8f acc[2][2] = {};

    for (int k0 = 0; k0 < K; k0 += 32) {
        async_ld16(sA0, &A[(blockM + arow0) * (size_t)lda + k0 + acol]);
        async_ld16(sA1, &A[(blockM + arow1) * (size_t)lda + k0 + acol]);
        async_ld16(sW0, &W[(size_t)(blockN + arow0) * ldw + k0 + acol]);
        wait_async0();
        __syncthreads();

        // A fragments (16x32 bf16, ISA lane/VGPR layout), two M sub-tiles
        Frag af[2], bf[2];
        #pragma unroll
        for (int mt = 0; mt < 2; ++mt) {
            #pragma unroll
            for (int v = 0; v < 8; ++v) {
                int kk = (v < 4 ? 2 * v : 2 * v + 8) + 8 * hi;
                af[mt].u[v] = *(const unsigned int*)
                    &sA[(waveM * 32 + mt * 16 + lr) * 32 + kk];
            }
        }
        // B fragments (32x16): lanes 0-15 K=0..7,16..23, lanes 16-31 K=8..15,24..31
        #pragma unroll
        for (int nt = 0; nt < 2; ++nt) {
            #pragma unroll
            for (int v = 0; v < 8; ++v) {
                int kk = (v < 4 ? 2 * v : 2 * v + 8) + 8 * hi;
                bf[nt].u[v] = *(const unsigned int*)
                    &sW[(waveN * 32 + nt * 16 + lr) * 32 + kk];
            }
        }
        #pragma unroll
        for (int mt = 0; mt < 2; ++mt)
            #pragma unroll
            for (int nt = 0; nt < 2; ++nt)
                acc[mt][nt] = __builtin_amdgcn_wmma_f32_16x16x32_bf16(
                    false, af[mt].v, false, bf[nt].v, (short)0,
                    acc[mt][nt], false, false);
        __syncthreads();
    }

    // C layout: VGPR r -> row r (lanes 0-15) / row r+8 (lanes 16-31), col = lane%16
    #pragma unroll
    for (int mt = 0; mt < 2; ++mt) {
        const size_t mbase = blockM + waveM * 32 + mt * 16 + hi * 8;
        #pragma unroll
        for (int nt = 0; nt < 2; ++nt) {
            const int n = blockN + waveN * 32 + nt * 16 + lr;
            #pragma unroll
            for (int r = 0; r < 8; ++r) {
                float  val = acc[mt][nt][r];
                size_t o   = (mbase + r) * (size_t)ldo + n;
                if (MODE == 0) {
                    outB[o] = (__bf16)val;
                } else if (MODE == 1) {
                    float xv = val + bias[n];
                    outF[o] = (xv > 20.f) ? xv : log1pf(__expf(xv));
                } else if (MODE == 2) {
                    outF[o] = val;
                    outB[o] = (__bf16)val;
                } else {
                    outF[o] = val + resid[o];
                }
            }
        }
    }
}

// ---------------------------------------------------------------------------
// Depthwise causal conv (D_CONV=4) + bias + SiLU. u = xz[:, 0:1024].
// ---------------------------------------------------------------------------
__global__ __launch_bounds__(256)
void conv_silu_kernel(const __bf16* __restrict__ xz, const float* __restrict__ cw,
                      const float* __restrict__ cb, __bf16* __restrict__ uc)
{
    size_t i  = (size_t)blockIdx.x * 256 + threadIdx.x;  // (b*L + l)*1024 + d
    int    d  = (int)(i & 1023);
    size_t bl = i >> 10;
    int    l  = (int)(bl & 4095);
    float acc = cb[d];
    #pragma unroll
    for (int k = 0; k < 4; ++k) {
        int ls = l - 3 + k;
        if (ls >= 0)
            acc += (float)xz[(bl - 3 + k) * 2048 + d] * cw[d * 4 + k];
    }
    float s = acc / (1.f + __expf(-acc));   // SiLU
    uc[i] = (__bf16)s;
}

// ---------------------------------------------------------------------------
// Selective scan, sequential over L. One thread per (b, d) channel carrying
// 16 f32 states. B/C vectors staged in LDS per step. Fused epilogue:
// y = scan_y + u*Dp, gated by SiLU(z), stored bf16 for the output GEMM.
// ---------------------------------------------------------------------------
__global__ __launch_bounds__(256)
void scan_kernel(const float* __restrict__ xdbl, const float* __restrict__ dtb,
                 const __bf16* __restrict__ uc, const __bf16* __restrict__ xz,
                 const float* __restrict__ Aneg, const float* __restrict__ Dp,
                 __bf16* __restrict__ yg)
{
    const int t = threadIdx.x;
    const int b = blockIdx.x >> 2;
    const int d = ((blockIdx.x & 3) << 8) + t;

    float h[DSTATE], a[DSTATE];
    #pragma unroll
    for (int n = 0; n < DSTATE; ++n) { h[n] = 0.f; a[n] = Aneg[d * DSTATE + n]; }
    const float dp = Dp[d];

    __shared__ float sB[DSTATE], sC[DSTATE];
    const size_t tokBase = (size_t)b * 4096;

    for (int l = 0; l < 4096; ++l) {
        size_t tok = tokBase + l;
        if (t < 32) {
            const float* xd = xdbl + tok * 64;
            if (t < 16) sB[t] = xd[32 + t];
            else        sC[t - 16] = xd[32 + t];
        }
        __syncthreads();

        size_t idx = tok * (size_t)DINNER + d;
        float dt = dtb[idx];
        float u  = (float)uc[idx];
        float du = dt * u;
        float y  = 0.f;
        #pragma unroll
        for (int n = 0; n < DSTATE; ++n) {
            float dA = __expf(dt * a[n]);
            h[n] = fmaf(dA, h[n], du * sB[n]);
            y    = fmaf(h[n], sC[n], y);
        }
        y = fmaf(u, dp, y);

        float z  = (float)xz[tok * 2048 + 1024 + d];
        float sg = z / (1.f + __expf(-z));
        yg[idx] = (__bf16)(y * sg);
        __syncthreads();
    }
}

// ---------------------------------------------------------------------------
extern "C" void kernel_launch(void* const* d_in, const int* in_sizes, int n_in,
                              void* d_out, int out_size, void* d_ws, size_t ws_size,
                              hipStream_t stream)
{
    (void)in_sizes; (void)n_in; (void)out_size; (void)ws_size;
    const float* x      = (const float*)d_in[0];
    const float* ln_g   = (const float*)d_in[1];
    const float* ln_b   = (const float*)d_in[2];
    const float* W_in   = (const float*)d_in[3];
    const float* conv_w = (const float*)d_in[4];
    const float* conv_b = (const float*)d_in[5];
    const float* W_xp   = (const float*)d_in[6];
    const float* W_dt   = (const float*)d_in[7];
    const float* b_dt   = (const float*)d_in[8];
    const float* A_log  = (const float*)d_in[9];
    const float* Dp     = (const float*)d_in[10];
    const float* W_out  = (const float*)d_in[11];
    float* out = (float*)d_out;

    char* w = (char*)d_ws;
    __bf16* xn    = (__bf16*)(w + OFF_XN);
    __bf16* xz    = (__bf16*)(w + OFF_XZ);
    __bf16* uc    = (__bf16*)(w + OFF_UC);
    float*  xdblF = (float*) (w + OFF_XDBLF);
    __bf16* xdblB = (__bf16*)(w + OFF_XDBLB);
    float*  dtf   = (float*) (w + OFF_DT);
    __bf16* yg    = (__bf16*)(w + OFF_YG);
    __bf16* bWin  = (__bf16*)(w + OFF_WIN);
    __bf16* bWxp  = (__bf16*)(w + OFF_WXP);
    __bf16* bWdt  = (__bf16*)(w + OFF_WDT);
    __bf16* bWout = (__bf16*)(w + OFF_WOUT);
    float*  Aneg  = (float*) (w + OFF_ANEG);

    // 1) weights -> bf16, A = -exp(A_log)
    prep_kernel<<<6592, 256, 0, stream>>>(W_in, W_xp, W_dt, W_out, A_log,
                                          bWin, bWxp, bWdt, bWout, Aneg);
    // 2) LayerNorm -> xn (bf16)
    layernorm_kernel<<<NTOK, 256, 0, stream>>>(x, ln_g, ln_b, xn);
    // 3) xz = xn @ W_in^T   [32768,2048], bf16 out
    wmma_gemm_kernel<0><<<dim3(32, 256), 256, 0, stream>>>(
        xn, bWin, DMODEL, DMODEL, DMODEL, 2048, nullptr, xz, nullptr, nullptr);
    // 4) depthwise conv4 + SiLU on u half -> uc (bf16)
    conv_silu_kernel<<<131072, 256, 0, stream>>>(xz, conv_w, conv_b, uc);
    // 5) x_dbl = uc @ W_xp^T [32768,64], f32 + bf16 out
    wmma_gemm_kernel<2><<<dim3(1, 256), 256, 0, stream>>>(
        uc, bWxp, DINNER, DINNER, DINNER, 64, xdblF, xdblB, nullptr, nullptr);
    // 6) dt = softplus(dt_r @ W_dt^T + b_dt) [32768,1024], f32 out
    wmma_gemm_kernel<1><<<dim3(16, 256), 256, 0, stream>>>(
        xdblB, bWdt, 64, 32, 32, DINNER, dtf, nullptr, b_dt, nullptr);
    // 7) selective scan + D-skip + SiLU gate -> yg (bf16)
    scan_kernel<<<32, 256, 0, stream>>>(xdblF, dtf, uc, xz, Aneg, Dp, yg);
    // 8) out = residual + yg @ W_out^T [32768,512], f32 out
    wmma_gemm_kernel<3><<<dim3(8, 256), 256, 0, stream>>>(
        yg, bWout, DINNER, DINNER, DINNER, DMODEL, out, nullptr, nullptr, x);
}
